// UBNeck_43636867727798
// MI455X (gfx1250) — matrix operand order repaired and checked by reading, not compiled, not run on hardware
//
#include <hip/hip_runtime.h>
#include <hip/hip_bf16.h>

// ---------------- types for WMMA fragments ----------------
typedef _Float16 v16h __attribute__((ext_vector_type(16)));
typedef _Float16 v8h  __attribute__((ext_vector_type(8)));
typedef float    v8f  __attribute__((ext_vector_type(8)));
typedef unsigned int u32x4 __attribute__((ext_vector_type(4)));
typedef int          i32x4 __attribute__((ext_vector_type(4)));
typedef int          i32x8 __attribute__((ext_vector_type(8)));

#if defined(__has_builtin)
#  if __has_builtin(__builtin_amdgcn_tensor_load_to_lds) && \
      __has_builtin(__builtin_amdgcn_s_wait_tensorcnt)
#    define USE_TDM 1
#  else
#    define USE_TDM 0
#  endif
#else
#  define USE_TDM 0
#endif

union HU { v16h v; v8h h[2]; };

// A row-major in LDS: A[o_row][Kt], k-contiguous.  a[e] needs
// k = 8*hi + (e&7) + 16*(e>>3)  -> two b128 loads at +8*hi and +8*hi+16.
// B stored as [s_row][Kt] (holds B[k][s]).  b[e] needs
// k = 8*hi + (e&7) + 16*(e>>3) -> two b128 loads at +8*hi and +8*hi+16.
__device__ __forceinline__ v8f wmma_f16(const _Float16* ap, const _Float16* bp, v8f acc) {
  HU A, B;
  A.h[0] = *(const v8h*)(ap);
  A.h[1] = *(const v8h*)(ap + 16);
  B.h[0] = *(const v8h*)(bp);
  B.h[1] = *(const v8h*)(bp + 16);
  return __builtin_amdgcn_wmma_f32_16x16x32_f16(false, A.v, false, B.v,
                                                (short)0, acc, false, false);
}

// TDM: async-load a dense 2D tile (rows x 128 f16, src row stride = strideElems)
// from global into LDS.  D# layout per CDNA5 ISA ch.8 (groups 0/1; 2/3 zero).
__device__ __forceinline__ void tdm_load_tile_f16(const _Float16* gsrc, void* ldsDst,
                                                  int rows, int strideElems) {
#if USE_TDM
  unsigned long long ga = (unsigned long long)(__UINTPTR_TYPE__)gsrc;
  unsigned int lds = (unsigned int)(__UINTPTR_TYPE__)ldsDst;  // addr[31:0] = LDS offset
  u32x4 g0;
  g0[0] = 1u;                                          // count=1 (valid user D#)
  g0[1] = lds;                                         // lds_addr
  g0[2] = (unsigned int)(ga & 0xFFFFFFFFu);            // global_addr[31:0]
  g0[3] = (unsigned int)((ga >> 32) & 0x01FFFFFFu)     // global_addr[56:32]
        | (2u << 30);                                  // type=2 ("image")
  unsigned int td0 = (unsigned int)strideElems;        // tensor_dim0
  unsigned int td1 = (unsigned int)rows;               // tensor_dim1
  i32x8 g1;
  g1[0] = (int)(1u << 16);                             // wg_mask=0, data_size=1 (2B)
  g1[1] = (int)((td0 & 0xFFFFu) << 16);                // tensor_dim0[15:0] @63:48
  g1[2] = (int)((td0 >> 16) | ((td1 & 0xFFFFu) << 16));// dim0[31:16], dim1[15:0]
  g1[3] = (int)((td1 >> 16) | (128u << 16));           // dim1[31:16], tile_dim0=128
  g1[4] = (int)td1;                                    // tile_dim1=rows, tile_dim2=0
  g1[5] = (int)td0;                                    // tensor_dim0_stride[31:0]
  g1[6] = 0;                                           // stride0[47:32], stride1[15:0]
  g1[7] = 0;
  i32x4 z4; z4[0] = 0; z4[1] = 0; z4[2] = 0; z4[3] = 0;
#if __clang_major__ >= 23
  i32x8 z8; z8[0]=0; z8[1]=0; z8[2]=0; z8[3]=0; z8[4]=0; z8[5]=0; z8[6]=0; z8[7]=0;
  __builtin_amdgcn_tensor_load_to_lds(g0, g1, z4, z4, z8, 0);
#else
  __builtin_amdgcn_tensor_load_to_lds(g0, g1, z4, z4, 0);
#endif
  __builtin_amdgcn_s_wait_tensorcnt(0);
#else
  (void)gsrc; (void)ldsDst; (void)rows; (void)strideElems;
#endif
}

__device__ __forceinline__ float block_reduce_max(float v, float* sm) {
  int t = threadIdx.x;
  sm[t] = v; __syncthreads();
  for (int s = 128; s > 0; s >>= 1) {
    if (t < s) sm[t] = fmaxf(sm[t], sm[t + s]);
    __syncthreads();
  }
  float r = sm[0]; __syncthreads();
  return r;
}

// ---------------- weight fake-quant prep ----------------
// w1: (128,32,1,1), per-channel (axis i) amax per o -> WT1[o][i] f16
__global__ __launch_bounds__(256) void prep_w1_kernel(const float* w1, _Float16* WT1) {
  __shared__ float sm[256];
  int o = blockIdx.x, t = threadIdx.x;
  float m = 0.f;
  for (int i = t; i < 128; i += 256) m = fmaxf(m, fabsf(w1[i * 32 + o]));
  m = block_reduce_max(m, sm);
  float scale = fmaxf(m, 1e-8f) / 127.0f;
  for (int i = t; i < 128; i += 256) {
    float q = fminf(fmaxf(rintf(w1[i * 32 + o] / scale), -127.f), 127.f) * scale;
    WT1[o * 128 + i] = (_Float16)q;
  }
}

// w2: (32,64,1,1), per-tensor amax -> WT2[o][i] f16 (64x32)
__global__ __launch_bounds__(256) void prep_w2_kernel(const float* w2, _Float16* WT2) {
  __shared__ float sm[256];
  int t = threadIdx.x;
  float m = 0.f;
  for (int idx = t; idx < 2048; idx += 256) m = fmaxf(m, fabsf(w2[idx]));
  m = block_reduce_max(m, sm);
  float scale = fmaxf(m, 1e-8f) / 127.0f;
  for (int idx = t; idx < 2048; idx += 256) {
    int i = idx >> 6, o = idx & 63;  // w2[i][o]
    float q = fminf(fmaxf(rintf(w2[idx] / scale), -127.f), 127.f) * scale;
    WT2[o * 32 + i] = (_Float16)q;
  }
}

// convT weights (CI,CO,4,4): per-o amax over (i,kh,kw); build per-parity
// packed weights Wp[pa*2+pb][o][i*4 + t*2 + s] = q(w[i][o][3-2t-pa][3-2s-pb])
template <int CI, int CO>
__global__ __launch_bounds__(256) void prep_convT_kernel(const float* w, _Float16* Wp) {
  __shared__ float sm[256];
  int o = blockIdx.x, t = threadIdx.x;
  float m = 0.f;
  for (int idx = t; idx < CI * 16; idx += 256) {
    int i = idx >> 4, r = idx & 15;
    m = fmaxf(m, fabsf(w[(i * CO + o) * 16 + r]));
  }
  m = block_reduce_max(m, sm);
  float scale = fmaxf(m, 1e-8f) / 127.0f;
  const int K = CI * 4;
  for (int idx = t; idx < 4 * CI * 4; idx += 256) {
    int pa = idx / (CI * 4);
    int rem = idx - pa * (CI * 4);
    int i = rem >> 2, ts = rem & 3;
    int tt = ts >> 1, sv = ts & 1;
    int a2 = pa >> 1, b2 = pa & 1;
    int kh = 3 - 2 * tt - a2;
    int kw = 3 - 2 * sv - b2;
    float wv = w[((i * CO + o) * 4 + kh) * 4 + kw];
    float q = fminf(fmaxf(rintf(wv / scale), -127.f), 127.f) * scale;
    Wp[((size_t)pa * CO + o) * K + i * 4 + ts] = (_Float16)q;
  }
}

// ---------------- 1x1 conv as WMMA GEMM ----------------
// out[n][o][s] = sum_i act[n][i][s] * W[o][i]; 64 spatial per block.
template <int CI, int CO>
__global__ __launch_bounds__(256) void gemm1x1_kernel(const float* actF, const _Float16* actH,
                                                      const _Float16* Wg, _Float16* outp,
                                                      int HWin) {
  __shared__ alignas(16) _Float16 Bt[64 * CI];
  __shared__ alignas(16) _Float16 Wt[CO * CI];
  int tid = threadIdx.x;
  int nb = HWin >> 6;
  int n = blockIdx.x / nb;
  int s0 = (blockIdx.x - n * nb) << 6;
  for (int idx = tid; idx < 64 * CI; idx += 256) {
    int i = idx >> 6, s = idx & 63;
    size_t g = ((size_t)n * CI + i) * HWin + s0 + s;
    float v = actF ? actF[g] : (float)actH[g];
    Bt[s * CI + i] = (_Float16)v;
  }
  for (int idx = tid; idx < CO * CI; idx += 256) Wt[idx] = Wg[idx];
  __syncthreads();
  int wave = tid >> 5, lane = tid & 31, l = lane & 15, hi = lane >> 4;
  const int TASKS = 4 * (CO / 16);
  for (int task = wave; task < TASKS; task += 8) {
    int st = task & 3, ot = task >> 2;
    v8f acc = {0.f, 0.f, 0.f, 0.f, 0.f, 0.f, 0.f, 0.f};
#pragma unroll
    for (int kk = 0; kk < CI / 32; ++kk) {
      acc = wmma_f16(&Wt[(ot * 16 + l) * CI + kk * 32 + 8 * hi],
                     &Bt[(st * 16 + l) * CI + kk * 32 + 8 * hi], acc);
    }
#pragma unroll
    for (int r = 0; r < 8; ++r) {
      int o = ot * 16 + r + 8 * hi;
      outp[((size_t)n * CO + o) * HWin + s0 + st * 16 + l] = (_Float16)acc[r];
    }
  }
}

// ---------------- 4x4 stride-2 pad-1 transposed conv as parity GEMMs -------
// out[n][o][2u+pa][2v+pb] = sum_{i,t,s} act[n][i][u+t-1+pa][v+s-1+pb]
//                                      * w[i][o][3-2t-pa][3-2s-pb]
// K = 4*CI, processed in chunks of 128; one block = (n, pa, pb, u), 64 v's.
// Weight chunk tile (CO x 128 f16, row stride K) is fetched with the
// Tensor Data Mover (async DMA, TENSORcnt) when available.
template <int CI, int CO>
__global__ __launch_bounds__(256) void convT_kernel(const float* actF, const _Float16* actH,
                                                    const _Float16* Wp, _Float16* outp) {
  __shared__ alignas(16) _Float16 Bt[64 * 128];
  __shared__ alignas(16) _Float16 Wt[CO * 128];
  const int Hin = 64, Win = 64;
  const int K = CI * 4;
  const int NCH = K / 128;
  int bid = blockIdx.x;
  int u = bid & 63; bid >>= 6;
  int pb = bid & 1; bid >>= 1;
  int pa = bid & 1; bid >>= 1;
  int n = bid;
  int tid = threadIdx.x;
  int wave = tid >> 5, lane = tid & 31, l = lane & 15, hi = lane >> 4;
  constexpr int NT = (4 * (CO / 16)) / 8;  // wmma tiles per wave (1 or 2)
  v8f acc[NT];
  const v8f vz = {0.f, 0.f, 0.f, 0.f, 0.f, 0.f, 0.f, 0.f};
#pragma unroll
  for (int j = 0; j < NT; ++j) acc[j] = vz;

  for (int ch = 0; ch < NCH; ++ch) {
    // im2col chunk: k-local 0..127 -> (i = ch*32 + k/4, t=(k>>1)&1, s=k&1)
    for (int idx = tid; idx < 64 * 128; idx += 256) {
      int k = idx >> 6, vv = idx & 63;
      int i = (ch << 5) + (k >> 2);
      int tt = (k >> 1) & 1, sv = k & 1;
      int row = u + tt - 1 + pa;
      int col = vv + sv - 1 + pb;
      float val = 0.f;
      if ((unsigned)row < 64u && (unsigned)col < 64u) {
        size_t g = (((size_t)n * CI + i) * Hin + row) * Win + col;
        val = actF ? actF[g] : (float)actH[g];
      }
      Bt[vv * 128 + k] = (_Float16)val;
    }
    const _Float16* wsrc = Wp + ((size_t)(pa * 2 + pb) * CO) * K + ch * 128;
    __builtin_prefetch(wsrc + 128, 0, 1);  // speculative: next K chunk
#if USE_TDM
    if (wave == 0) {
      // one wave issues the DMA; barrier below publishes the tile
      tdm_load_tile_f16(wsrc, &Wt[0], CO, K);
    }
#else
    for (int idx = tid; idx < CO * 128; idx += 256) {
      int o = idx >> 7, kl = idx & 127;
      Wt[idx] = wsrc[(size_t)o * K + kl];
    }
#endif
    __syncthreads();
#pragma unroll
    for (int j = 0; j < NT; ++j) {
      int task = wave + j * 8;
      int st = task & 3, ot = task >> 2;
#pragma unroll
      for (int kk = 0; kk < 4; ++kk) {
        acc[j] = wmma_f16(&Wt[(ot * 16 + l) * 128 + kk * 32 + 8 * hi],
                          &Bt[(st * 16 + l) * 128 + kk * 32 + 8 * hi], acc[j]);
      }
    }
    __syncthreads();
  }
  int p = 2 * u + pa;
#pragma unroll
  for (int j = 0; j < NT; ++j) {
    int task = wave + j * 8;
    int st = task & 3, ot = task >> 2;
#pragma unroll
    for (int r = 0; r < 8; ++r) {
      int o = ot * 16 + r + 8 * hi;
      int q = ((st * 16 + l) << 1) + pb;
      outp[(((size_t)n * CO + o) * 128 + p) * 128 + q] = (_Float16)acc[j][r];
    }
  }
}

// ---------------- batchnorm stats / apply / final -------------------------
__global__ __launch_bounds__(256) void bn_stats_kernel(const _Float16* h, int C, int NN, int HW,
                                                       float* mean, float* rstd) {
  __shared__ float sA[256];
  __shared__ float sB[256];
  int c = blockIdx.x, t = threadIdx.x;
  float sum = 0.f, sq = 0.f;
  int cnt = NN * HW;
  for (int idx = t; idx < cnt; idx += 256) {
    int nn = idx / HW;
    int s = idx - nn * HW;
    float v = (float)h[((size_t)nn * C + c) * HW + s];
    sum += v; sq += v * v;
  }
  sA[t] = sum; sB[t] = sq; __syncthreads();
  for (int st = 128; st > 0; st >>= 1) {
    if (t < st) { sA[t] += sA[t + st]; sB[t] += sB[t + st]; }
    __syncthreads();
  }
  if (t == 0) {
    float mu = sA[0] / (float)cnt;
    float var = sB[0] / (float)cnt - mu * mu;
    mean[c] = mu;
    rstd[c] = rsqrtf(var + 1e-5f);
  }
}

__global__ __launch_bounds__(256) void bn_qrelu_kernel(_Float16* h, int C, int HW,
                                                       const float* mean, const float* rstd,
                                                       const float* g, const float* bb,
                                                       const float* sPtr, int total) {
  int idx = blockIdx.x * 256 + threadIdx.x;
  if (idx >= total) return;
  int c = (idx / HW) % C;
  float v = (float)h[idx];
  float y = (v - mean[c]) * rstd[c] * g[c] + bb[c];
  float r = fmaxf(y, 0.f);
  float step = sPtr[0] * (1.0f / 255.0f);
  float q = fminf(fmaxf(rintf(r / step), 0.f), 255.f) * step;
  h[idx] = (_Float16)q;
}

__global__ __launch_bounds__(256) void final_kernel(const _Float16* h3, const _Float16* sc,
                                                    const float* mean, const float* rstd,
                                                    const float* g, const float* bb,
                                                    const float* sPtr, float* out, int total) {
  int idx = blockIdx.x * 256 + threadIdx.x;
  if (idx >= total) return;
  int c = (idx / 16384) % 64;
  float v = (float)h3[idx];
  float y = (v - mean[c]) * rstd[c] * g[c] + bb[c] + (float)sc[idx];
  float r = fmaxf(y, 0.f);
  float step = sPtr[0] * (1.0f / 255.0f);
  out[idx] = fminf(fmaxf(rintf(r / step), 0.f), 255.f) * step;
}

// ---------------- workspace layout (bytes) ----------------
#define OFF_WT1  ((size_t)0)          //  8192  B : WT1 [32][128] f16
#define OFF_WT2  ((size_t)8192)       //  4096  B : WT2 [64][32]  f16
#define OFF_WPM  ((size_t)16384)      // 32768  B : Wp main [4][32][128] f16
#define OFF_WPS  ((size_t)49152)      // 262144 B : Wp sc   [4][64][512] f16
#define OFF_STAT ((size_t)311296)     //  4096  B : bn stats (256 floats)
#define OFF_H1   ((size_t)315392)     //  8 MiB : h1  [32][32][4096]  f16
#define OFF_H2   ((size_t)8704000)    // 32 MiB : h2  [32][32][16384] f16
#define OFF_H3   ((size_t)42258432)   // 64 MiB : h3  [32][64][16384] f16
#define OFF_SC   ((size_t)109367296)  // 64 MiB : sc  [32][64][16384] f16
// total ~176.5 MiB

extern "C" void kernel_launch(void* const* d_in, const int* in_sizes, int n_in,
                              void* d_out, int out_size, void* d_ws, size_t ws_size,
                              hipStream_t stream) {
  const float* x   = (const float*)d_in[0];
  const float* w1  = (const float*)d_in[1];
  const float* wum = (const float*)d_in[2];
  const float* w2  = (const float*)d_in[3];
  const float* wus = (const float*)d_in[4];
  const float* g1  = (const float*)d_in[5];
  const float* b1  = (const float*)d_in[6];
  const float* g2  = (const float*)d_in[7];
  const float* b2  = (const float*)d_in[8];
  const float* g3  = (const float*)d_in[9];
  const float* b3  = (const float*)d_in[10];
  const float* s1  = (const float*)d_in[11];
  const float* s2  = (const float*)d_in[12];
  const float* s3  = (const float*)d_in[13];
  float* out = (float*)d_out;
  char* ws = (char*)d_ws;

  _Float16* WT1 = (_Float16*)(ws + OFF_WT1);
  _Float16* WT2 = (_Float16*)(ws + OFF_WT2);
  _Float16* WPM = (_Float16*)(ws + OFF_WPM);
  _Float16* WPS = (_Float16*)(ws + OFF_WPS);
  float* st = (float*)(ws + OFF_STAT);
  float* mean1 = st +   0; float* rstd1 = st +  32;
  float* mean2 = st +  64; float* rstd2 = st +  96;
  float* mean3 = st + 128; float* rstd3 = st + 192;
  _Float16* H1 = (_Float16*)(ws + OFF_H1);
  _Float16* H2 = (_Float16*)(ws + OFF_H2);
  _Float16* H3 = (_Float16*)(ws + OFF_H3);
  _Float16* SC = (_Float16*)(ws + OFF_SC);

  dim3 blk(256);

  // weight fake-quant prep (tiny)
  prep_w1_kernel<<<dim3(32), blk, 0, stream>>>(w1, WT1);
  prep_w2_kernel<<<dim3(1), blk, 0, stream>>>(w2, WT2);
  prep_convT_kernel<32, 32><<<dim3(32), blk, 0, stream>>>(wum, WPM);
  prep_convT_kernel<128, 64><<<dim3(64), blk, 0, stream>>>(wus, WPS);

  // stage 1: 1x1 conv (K=128 -> 32) over 64x64
  gemm1x1_kernel<128, 32><<<dim3(32 * 64), blk, 0, stream>>>(x, nullptr, WT1, H1, 4096);
  bn_stats_kernel<<<dim3(32), blk, 0, stream>>>(H1, 32, 32, 4096, mean1, rstd1);
  bn_qrelu_kernel<<<dim3(16384), blk, 0, stream>>>(H1, 32, 4096, mean1, rstd1, g1, b1, s1,
                                                   32 * 32 * 4096);

  // stage 2: convT 4x4 s2 p1 (32 -> 32), 64x64 -> 128x128
  convT_kernel<32, 32><<<dim3(8192), blk, 0, stream>>>(nullptr, H1, WPM, H2);
  bn_stats_kernel<<<dim3(32), blk, 0, stream>>>(H2, 32, 32, 16384, mean2, rstd2);
  bn_qrelu_kernel<<<dim3(65536), blk, 0, stream>>>(H2, 32, 16384, mean2, rstd2, g2, b2, s2,
                                                   32 * 32 * 16384);

  // stage 3: 1x1 conv (K=32 -> 64) over 128x128
  gemm1x1_kernel<32, 64><<<dim3(32 * 256), blk, 0, stream>>>(nullptr, H2, WT2, H3, 16384);

  // shortcut: convT 4x4 s2 p1 (128 -> 64) on x
  convT_kernel<128, 64><<<dim3(8192), blk, 0, stream>>>(x, nullptr, WPS, SC);

  // BN3 on h3, add shortcut, final quant-ReLU -> f32 out
  bn_stats_kernel<<<dim3(64), blk, 0, stream>>>(H3, 64, 32, 16384, mean3, rstd3);
  final_kernel<<<dim3(131072), blk, 0, stream>>>(H3, SC, mean3, rstd3, g3, b3, s3, out,
                                                 32 * 64 * 16384);
}
